// SGAttentionStem_13778255086242
// MI455X (gfx1250) — matrix operand order repaired and checked by reading, not compiled, not run on hardware
//
// CDNA5 (gfx1250) fused segment-attention stem.
// Memory-bound (~0.9 GB traffic @ 23.3 TB/s); all dense linears use
// v_wmma_f32_16x16x32_bf16 (fp32 accumulate); softmax/LN in fp32.
// Weights are pre-packed into the WMMA B-fragment register image so
// fragment loads are 2x global_load_b128 (no strided gathers, no spills).
#include <hip/hip_runtime.h>

#define NPIX      65536
#define DIMS      64
#define FFN_DIMS  192
#define NENTRIES  1048576
#define LN_EPS    1e-6f
#define SM_EPS    1.1920929e-7f
#define NEG_INF   (-__builtin_inff())

typedef __bf16 bf16;
typedef __attribute__((ext_vector_type(16))) __bf16 v16bf;
typedef __attribute__((ext_vector_type(8)))  __bf16 v8bf;
typedef __attribute__((ext_vector_type(8)))  float  v8f;

__device__ __forceinline__ void wmma_bf16(const v16bf& a, const v16bf& b, v8f& c) {
  c = __builtin_amdgcn_wmma_f32_16x16x32_bf16(false, a, false, b, (short)0, c, false, false);
}

// ---- fragment loaders (ISA 7.12.2 layouts) --------------------------------
// A (16x32 bf16): lane l holds row (l&15); halves 0..7 = K {sel*8..sel*8+7},
// halves 8..15 = K {16+sel*8 ..}, sel = l>=16.  => two contiguous 8-elem runs.
__device__ __forceinline__ v16bf afrag_f32(const float* __restrict__ base, int stride,
                                           int lane, int nrows) {
  int r = lane & 15, sel = (lane >> 4) & 1;
  v16bf a;
  if (r < nrows) {
    const float* p = base + r * stride + sel * 8;
#pragma unroll
    for (int h = 0; h < 8; ++h) { a[h] = (__bf16)p[h]; a[h + 8] = (__bf16)p[h + 16]; }
  } else {
#pragma unroll
    for (int h = 0; h < 16; ++h) a[h] = (__bf16)0.0f;
  }
  return a;
}

__device__ __forceinline__ v16bf afrag_bf16(const bf16* __restrict__ base, int stride,
                                            int lane, int nrows) {
  int r = lane & 15, sel = (lane >> 4) & 1;
  if (r < nrows) {
    const bf16* p = base + r * stride + sel * 8;
    v8bf lo = *(const v8bf*)p;
    v8bf hi = *(const v8bf*)(p + 16);
    return __builtin_shufflevector(lo, hi, 0,1,2,3,4,5,6,7,8,9,10,11,12,13,14,15);
  }
  v16bf z;
#pragma unroll
  for (int h = 0; h < 16; ++h) z[h] = (__bf16)0.0f;
  return z;
}

// B (32x16 bf16): lane l holds column (l&15); same K order as A (sel*8.., 16+sel*8..).
// Slow gather form (used once, in the pack kernel).
__device__ __forceinline__ v16bf bfrag_f32(const float* __restrict__ W, int N,
                                           int k0, int n0, int lane) {
  int n = n0 + (lane & 15);
  const float* p = W + (size_t)(k0 + ((lane >> 4) & 1) * 8) * N + n;
  v16bf b;
#pragma unroll
  for (int h = 0; h < 16; ++h) b[h] = (__bf16)p[(size_t)(h + (h >> 3) * 8) * N];
  return b;
}

// Packed fragment: 512 bf16 per fragment; lane l owns 16 contiguous halves.
__device__ __forceinline__ v16bf bfrag_packed(const bf16* __restrict__ P, int frag, int lane) {
  const bf16* p = P + ((size_t)frag << 9) + (lane << 4);
  v8bf lo = *(const v8bf*)p;
  v8bf hi = *(const v8bf*)(p + 8);
  return __builtin_shufflevector(lo, hi, 0,1,2,3,4,5,6,7,8,9,10,11,12,13,14,15);
}

__device__ __forceinline__ float gelu_tanh(float x) {
  const float c = 0.7978845608028654f;
  float t = tanhf(c * (x + 0.044715f * x * x * x));
  return 0.5f * x * (1.0f + t);
}

// LN over 64 features; C layout: lane l holds (m = rr + 8*(l>=16), n = l&15).
__device__ __forceinline__ void ln_store(float val[4][8], const float* __restrict__ g,
                                         const float* __restrict__ b,
                                         float* __restrict__ out, int p0, int lane) {
  int sel = (lane >> 4) & 1, n = lane & 15;
  float gv[4], bv[4];
#pragma unroll
  for (int nt = 0; nt < 4; ++nt) { gv[nt] = g[nt * 16 + n]; bv[nt] = b[nt * 16 + n]; }
#pragma unroll
  for (int rr = 0; rr < 8; ++rr) {
    float s = val[0][rr] + val[1][rr] + val[2][rr] + val[3][rr];
    s += __shfl_xor(s, 1); s += __shfl_xor(s, 2); s += __shfl_xor(s, 4); s += __shfl_xor(s, 8);
    float mean = s * (1.0f / 64.0f);
    float vs = 0.f;
#pragma unroll
    for (int nt = 0; nt < 4; ++nt) { float d = val[nt][rr] - mean; vs += d * d; }
    vs += __shfl_xor(vs, 1); vs += __shfl_xor(vs, 2); vs += __shfl_xor(vs, 4); vs += __shfl_xor(vs, 8);
    float rstd = rsqrtf(vs * (1.0f / 64.0f) + LN_EPS);
    int m = rr + sel * 8;
#pragma unroll
    for (int nt = 0; nt < 4; ++nt)
      out[(size_t)(p0 + m) * DIMS + nt * 16 + n] = (val[nt][rr] - mean) * rstd * gv[nt] + bv[nt];
  }
}

// ---- kernels ---------------------------------------------------------------
__global__ void k_zero(float* __restrict__ p, size_t n) {
  size_t i = (size_t)blockIdx.x * blockDim.x + threadIdx.x;
  size_t step = (size_t)gridDim.x * blockDim.x;
  for (; i < n; i += step) p[i] = 0.0f;
}

// Pack f32 weight [K,N] into bf16 B-fragment images: frag = ks*(N/16)+nt.
__global__ void k_pack(const float* __restrict__ W, int N, bf16* __restrict__ out) {
  int lane = threadIdx.x;
  int frag = blockIdx.x;
  int ntiles = N >> 4;
  int ks = frag / ntiles, nt = frag - ks * ntiles;
  v16bf b = bfrag_f32(W, N, ks * 32, nt * 16, lane);
  bf16* p = out + ((size_t)frag << 9) + (lane << 4);
  *(v8bf*)p       = __builtin_shufflevector(b, b, 0,1,2,3,4,5,6,7);
  *(v8bf*)(p + 8) = __builtin_shufflevector(b, b, 8,9,10,11,12,13,14,15);
}

// ctx[e] = (gene_emb[idx[e]] + cnt_emb[digitize(cnt[e])]) @ Wc + bc  -> bf16
__global__ void k_ctx(const int* __restrict__ indices, const int* __restrict__ cnts,
                      const float* __restrict__ gene_emb, const float* __restrict__ cnt_emb,
                      const bf16* __restrict__ pkW, const float* __restrict__ bc,
                      bf16* __restrict__ ctx) {
  int lane = threadIdx.x;
  int n = lane & 15, sel = (lane >> 4) & 1;
  v16bf B[2][4];
  float bias[4];
#pragma unroll
  for (int ks = 0; ks < 2; ++ks)
#pragma unroll
    for (int nt = 0; nt < 4; ++nt) B[ks][nt] = bfrag_packed(pkW, ks * 4 + nt, lane);
#pragma unroll
  for (int nt = 0; nt < 4; ++nt) bias[nt] = bc[nt * 16 + n];

#pragma unroll 1
  for (int tile = blockIdx.x; tile < NENTRIES / 16; tile += gridDim.x) {
    int e0 = tile * 16;
    int r = lane & 15;
    int idx = indices[e0 + r];
    int cnt = cnts[e0 + r];
    int dig = (cnt >= 1) + (cnt >= 2) + (cnt >= 4) + (cnt >= 8) +
              (cnt >= 16) + (cnt >= 32) + (cnt >= 64);
    const float* g  = gene_emb + (size_t)idx * DIMS;
    const float* ce = cnt_emb + (size_t)dig * DIMS;
    v8f C[4] = {};
#pragma unroll
    for (int ks = 0; ks < 2; ++ks) {
      int kb = ks * 32 + sel * 8;
      v16bf a;
#pragma unroll
      for (int h = 0; h < 8; ++h) {
        a[h]     = (__bf16)(g[kb + h]      + ce[kb + h]);
        a[h + 8] = (__bf16)(g[kb + 16 + h] + ce[kb + 16 + h]);
      }
#pragma unroll
      for (int nt = 0; nt < 4; ++nt) wmma_bf16(a, B[ks][nt], C[nt]);
    }
#pragma unroll
    for (int nt = 0; nt < 4; ++nt)
#pragma unroll
      for (int rr = 0; rr < 8; ++rr)
        ctx[(size_t)(e0 + rr + sel * 8) * DIMS + nt * 16 + n] = (__bf16)(C[nt][rr] + bias[nt]);
  }
}

// q = (x @ Wq + bq) * 0.25
__global__ void k_q(const float* __restrict__ x, const bf16* __restrict__ pkW,
                    const float* __restrict__ bq, float* __restrict__ q) {
  int lane = threadIdx.x;
  int n = lane & 15, sel = (lane >> 4) & 1;
  v16bf B[2][4];
  float bias[4];
#pragma unroll
  for (int ks = 0; ks < 2; ++ks)
#pragma unroll
    for (int nt = 0; nt < 4; ++nt) B[ks][nt] = bfrag_packed(pkW, ks * 4 + nt, lane);
#pragma unroll
  for (int nt = 0; nt < 4; ++nt) bias[nt] = bq[nt * 16 + n];

#pragma unroll 1
  for (int tile = blockIdx.x; tile < NPIX / 16; tile += gridDim.x) {
    int p0 = tile * 16;
    v8f C[4] = {};
#pragma unroll
    for (int ks = 0; ks < 2; ++ks) {
      v16bf a = afrag_f32(x + (size_t)p0 * DIMS + ks * 32, DIMS, lane, 16);
#pragma unroll
      for (int nt = 0; nt < 4; ++nt) wmma_bf16(a, B[ks][nt], C[nt]);
    }
#pragma unroll
    for (int nt = 0; nt < 4; ++nt)
#pragma unroll
      for (int rr = 0; rr < 8; ++rr)
        q[(size_t)(p0 + rr + sel * 8) * DIMS + nt * 16 + n] = 0.25f * (C[nt][rr] + bias[nt]);
  }
}

// Fused k/v projection + ragged online-softmax attention for 16 pixels/block.
__global__ void __launch_bounds__(128)
k_attn(const int* __restrict__ indptr, const bf16* __restrict__ ctx,
       const float* __restrict__ qg,
       const bf16* __restrict__ pkK, const float* __restrict__ kb,
       const bf16* __restrict__ pkV, const float* __restrict__ vb,
       float* __restrict__ res) {
  __shared__ float w_lds[64][4];
  __shared__ float v_lds[64][64];
  __shared__ float m_st[16][4], s_st[16][4];
  __shared__ float acc_st[16][4][16];
  __shared__ int   ip[17];

  int t = threadIdx.x, lane = t & 31, wave = t >> 5;
  int p0 = blockIdx.x * 16;
  if (t < 17) ip[t] = indptr[p0 + t];
  if (t < 64) { m_st[t >> 2][t & 3] = NEG_INF; s_st[t >> 2][t & 3] = 0.f; }
  for (int i = t; i < 16 * 4 * 16; i += 128) ((float*)acc_st)[i] = 0.f;

  int n = lane & 15, sel = (lane >> 4) & 1;
  v16bf BK[2][4], BV[2][4];
  float kbias[4], vbias[4];
#pragma unroll
  for (int ks = 0; ks < 2; ++ks)
#pragma unroll
    for (int nt = 0; nt < 4; ++nt) {
      BK[ks][nt] = bfrag_packed(pkK, ks * 4 + nt, lane);
      BV[ks][nt] = bfrag_packed(pkV, ks * 4 + nt, lane);
    }
#pragma unroll
  for (int nt = 0; nt < 4; ++nt) { kbias[nt] = kb[nt * 16 + n]; vbias[nt] = vb[nt * 16 + n]; }
  __syncthreads();

  int eb = ip[0], ee = ip[16];
#pragma unroll 1
  for (int c0 = eb; c0 < ee; c0 += 64) {
    // -------- phase A: k,v via WMMA; raw logits + v into LDS --------
    int se0 = c0 + wave * 16;
    int nrows = ee - se0; nrows = nrows > 16 ? 16 : nrows;
    if (nrows > 0) {
      if (lane == 0 && c0 + 64 < ee)
        __builtin_prefetch(ctx + (size_t)(c0 + 64) * DIMS, 0, 1);
      v8f CK[4] = {}, CV[4] = {};
#pragma unroll
      for (int ks = 0; ks < 2; ++ks) {
        v16bf a = afrag_bf16(ctx + (size_t)se0 * DIMS + ks * 32, DIMS, lane, nrows);
#pragma unroll
        for (int nt = 0; nt < 4; ++nt) {
          wmma_bf16(a, BK[ks][nt], CK[nt]);
          wmma_bf16(a, BV[ks][nt], CV[nt]);
        }
      }
      int pl[8];
      {
        int p = 0;
#pragma unroll
        for (int rr = 0; rr < 8; ++rr) {
          int e = se0 + rr + sel * 8;
          while (p < 15 && ip[p + 1] <= e) ++p;
          pl[rr] = p;
        }
      }
#pragma unroll
      for (int nt = 0; nt < 4; ++nt)
#pragma unroll
        for (int rr = 0; rr < 8; ++rr) {
          int m = rr + sel * 8;
          float kk = CK[nt][rr] + kbias[nt];
          float qq = qg[(size_t)(p0 + pl[rr]) * DIMS + nt * 16 + n];
          float prod = kk * qq;
          prod += __shfl_xor(prod, 1);
          prod += __shfl_xor(prod, 2);
          prod += __shfl_xor(prod, 4);
          prod += __shfl_xor(prod, 8);
          if (n == 0) w_lds[wave * 16 + m][nt] = (se0 + m < ee) ? prod : NEG_INF;
          v_lds[wave * 16 + m][nt * 16 + n] = CV[nt][rr] + vbias[nt];
        }
    }
    __syncthreads();
    // -------- phase B: online softmax merge; worker = (pixel, head, half) ---
    {
      int p = t >> 3, hh = (t >> 1) & 3, g = t & 1;
      int es = ip[p] > c0 ? ip[p] : c0;
      int ef = ip[p + 1] < c0 + 64 ? ip[p + 1] : c0 + 64;
      if (es < ef) {
        float mc = NEG_INF;
        for (int e = es; e < ef; ++e) mc = fmaxf(mc, w_lds[e - c0][hh]);
        float sc = 0.f, accc[8] = {0, 0, 0, 0, 0, 0, 0, 0};
        for (int e = es; e < ef; ++e) {
          float wv = expf(w_lds[e - c0][hh] - mc);
          sc += wv;
          const float* vp = &v_lds[e - c0][hh * 16 + g * 8];
#pragma unroll
          for (int f = 0; f < 8; ++f) accc[f] += wv * vp[f];
        }
        float mo = m_st[p][hh];
        float mn = fmaxf(mo, mc);
        float fo = expf(mo - mn);   // mo==-inf -> 0
        float fn = expf(mc - mn);
#pragma unroll
        for (int f = 0; f < 8; ++f)
          acc_st[p][hh][g * 8 + f] = acc_st[p][hh][g * 8 + f] * fo + accc[f] * fn;
        if (g == 0) { m_st[p][hh] = mn; s_st[p][hh] = s_st[p][hh] * fo + sc * fn; }
      }
    }
    __syncthreads();
  }
  // -------- finalize: res = acc / (s + EPS)  (0 if no entries) --------
  {
    int p = t >> 3, hh = (t >> 1) & 3, g = t & 1;
    float inv = 0.f;
    if (m_st[p][hh] != NEG_INF) inv = 1.f / (s_st[p][hh] + SM_EPS);
#pragma unroll
    for (int f = 0; f < 8; ++f)
      res[(size_t)(p0 + p) * DIMS + hh * 16 + g * 8 + f] = acc_st[p][hh][g * 8 + f] * inv;
  }
}

// x = LN(x + res @ Wo + bo)
__global__ void k_oln(const float* __restrict__ resb, const bf16* __restrict__ pkW,
                      const float* __restrict__ bo, const float* __restrict__ lg,
                      const float* __restrict__ lb, float* __restrict__ x) {
  int lane = threadIdx.x;
  int n = lane & 15, sel = (lane >> 4) & 1;
  v16bf B[2][4];
  float bias[4];
#pragma unroll
  for (int ks = 0; ks < 2; ++ks)
#pragma unroll
    for (int nt = 0; nt < 4; ++nt) B[ks][nt] = bfrag_packed(pkW, ks * 4 + nt, lane);
#pragma unroll
  for (int nt = 0; nt < 4; ++nt) bias[nt] = bo[nt * 16 + n];

#pragma unroll 1
  for (int tile = blockIdx.x; tile < NPIX / 16; tile += gridDim.x) {
    int p0 = tile * 16;
    v8f C[4] = {};
#pragma unroll
    for (int ks = 0; ks < 2; ++ks) {
      v16bf a = afrag_f32(resb + (size_t)p0 * DIMS + ks * 32, DIMS, lane, 16);
#pragma unroll
      for (int nt = 0; nt < 4; ++nt) wmma_bf16(a, B[ks][nt], C[nt]);
    }
    float val[4][8];
#pragma unroll
    for (int nt = 0; nt < 4; ++nt)
#pragma unroll
      for (int rr = 0; rr < 8; ++rr) {
        int m = rr + sel * 8;
        val[nt][rr] = C[nt][rr] + bias[nt] + x[(size_t)(p0 + m) * DIMS + nt * 16 + n];
      }
    ln_store(val, lg, lb, x, p0, lane);
  }
}

// h = gelu(x @ W1 + b1)  (bf16) ; grid.y = 3 feature groups of 64
__global__ void k_ffn1(const float* __restrict__ x, const bf16* __restrict__ pkW,
                       const float* __restrict__ b1, bf16* __restrict__ h) {
  int lane = threadIdx.x;
  int n = lane & 15, sel = (lane >> 4) & 1;
  int fg = blockIdx.y;
  int nbase = fg * 64;
  v16bf B[2][4];
  float bias[4];
#pragma unroll
  for (int ks = 0; ks < 2; ++ks)
#pragma unroll
    for (int nt = 0; nt < 4; ++nt)
      B[ks][nt] = bfrag_packed(pkW, ks * 12 + fg * 4 + nt, lane);
#pragma unroll
  for (int nt = 0; nt < 4; ++nt) bias[nt] = b1[nbase + nt * 16 + n];

#pragma unroll 1
  for (int tile = blockIdx.x; tile < NPIX / 16; tile += gridDim.x) {
    int p0 = tile * 16;
    v8f C[4] = {};
#pragma unroll
    for (int ks = 0; ks < 2; ++ks) {
      v16bf a = afrag_f32(x + (size_t)p0 * DIMS + ks * 32, DIMS, lane, 16);
#pragma unroll
      for (int nt = 0; nt < 4; ++nt) wmma_bf16(a, B[ks][nt], C[nt]);
    }
#pragma unroll
    for (int nt = 0; nt < 4; ++nt)
#pragma unroll
      for (int rr = 0; rr < 8; ++rr)
        h[(size_t)(p0 + rr + sel * 8) * FFN_DIMS + nbase + nt * 16 + n] =
            (__bf16)gelu_tanh(C[nt][rr] + bias[nt]);
  }
}

// x_out = LN(x_in + h @ W2 + b2); B-fragments streamed from packed buffer
// per k-step to keep live VGPRs low (K=192 would need 192 resident VGPRs).
__global__ void k_ffn2(const bf16* __restrict__ h, const bf16* __restrict__ pkW,
                       const float* __restrict__ b2, const float* __restrict__ lg,
                       const float* __restrict__ lb, const float* __restrict__ x_in,
                       float* __restrict__ x_out) {
  int lane = threadIdx.x;
  int n = lane & 15, sel = (lane >> 4) & 1;
  float bias[4];
#pragma unroll
  for (int nt = 0; nt < 4; ++nt) bias[nt] = b2[nt * 16 + n];

#pragma unroll 1
  for (int tile = blockIdx.x; tile < NPIX / 16; tile += gridDim.x) {
    int p0 = tile * 16;
    v8f C[4] = {};
#pragma unroll 1
    for (int ks = 0; ks < 6; ++ks) {
      v16bf a = afrag_bf16(h + (size_t)p0 * FFN_DIMS + ks * 32, FFN_DIMS, lane, 16);
#pragma unroll
      for (int nt = 0; nt < 4; ++nt) {
        v16bf b = bfrag_packed(pkW, ks * 4 + nt, lane);
        wmma_bf16(a, b, C[nt]);
      }
    }
    float val[4][8];
#pragma unroll
    for (int nt = 0; nt < 4; ++nt)
#pragma unroll
      for (int rr = 0; rr < 8; ++rr) {
        int m = rr + sel * 8;
        val[nt][rr] = C[nt][rr] + bias[nt] + x_in[(size_t)(p0 + m) * DIMS + nt * 16 + n];
      }
    ln_store(val, lg, lb, x_out, p0, lane);
  }
}

// ---- launch ----------------------------------------------------------------
extern "C" void kernel_launch(void* const* d_in, const int* in_sizes, int n_in,
                              void* d_out, int out_size, void* d_ws, size_t ws_size,
                              hipStream_t stream) {
  auto FP = [&](int i) { return (const float*)d_in[i]; };
  auto IP = [&](int i) { return (const int*)d_in[i]; };

  const int *indices, *cnts, *indptr;
  const float *gene, *cemb, *cw, *cb;
  struct LP { const float *qw,*qb,*kw,*kb,*vw,*vb,*ow,*ob,*l1g,*l1b,*f1w,*f1b,*f2w,*f2b,*l2g,*l2b; } L[3];

  bool sortedConv = (n_in > 3 && in_sizes[3] == 512);  // jax sorted-key flatten
  if (!sortedConv) {  // insertion order
    indices = IP(0); cnts = IP(1); indptr = IP(2);
    gene = FP(3); cemb = FP(4); cw = FP(5); cb = FP(6);
    for (int l = 0; l < 3; ++l) {
      int b = 7 + 16 * l;
      L[l] = { FP(b+0),FP(b+1),FP(b+2),FP(b+3),FP(b+4),FP(b+5),FP(b+6),FP(b+7),
               FP(b+8),FP(b+9),FP(b+10),FP(b+11),FP(b+12),FP(b+13),FP(b+14),FP(b+15) };
    }
  } else {            // sorted keys: cnts,indices,indptr | cnt_emb,ctx.b,ctx.w,gene,layers
    cnts = IP(0); indices = IP(1); indptr = IP(2);
    cemb = FP(3); cb = FP(4); cw = FP(5); gene = FP(6);
    for (int l = 0; l < 3; ++l) {
      int b = 7 + 16 * l;
      L[l].f1b = FP(b+0);  L[l].f1w = FP(b+1);  L[l].f2b = FP(b+2);  L[l].f2w = FP(b+3);
      L[l].kb  = FP(b+4);  L[l].kw  = FP(b+5);  L[l].l1b = FP(b+6);  L[l].l1g = FP(b+7);
      L[l].l2b = FP(b+8);  L[l].l2g = FP(b+9);  L[l].ob  = FP(b+10); L[l].ow  = FP(b+11);
      L[l].qb  = FP(b+12); L[l].qw  = FP(b+13); L[l].vb  = FP(b+14); L[l].vw  = FP(b+15);
    }
  }

  // workspace layout (all 256B aligned)
  char* ws = (char*)d_ws;
  bf16*  ctx  = (bf16*)(ws);                                  // 1M x 64 bf16 = 128 MiB
  float* x    = (float*)(ws + (size_t)134217728);             // 64K x 64 f32 = 16 MiB
  float* q    = (float*)(ws + (size_t)150994944);             // 16 MiB
  float* resb = (float*)(ws + (size_t)167772160);             // 16 MiB
  bf16*  hbuf = (bf16*)(ws + (size_t)184549376);              // 64K x 192 bf16 = 24 MiB
  bf16*  pkbase = (bf16*)(ws + (size_t)209715200);            // packed weights ~250 KB
  (void)ws_size; (void)out_size;

  // packed fragment buffers: 64x64 -> 8 frags (4096 bf16); 64x192 / 192x64 -> 24 frags
  size_t off = 0;
  bf16* pk_ctx = pkbase + off; off += 8 * 512;
  bf16 *pk_q[3], *pk_k[3], *pk_v[3], *pk_o[3], *pk_f1[3], *pk_f2[3];
  for (int l = 0; l < 3; ++l) {
    pk_q[l]  = pkbase + off; off += 8 * 512;
    pk_k[l]  = pkbase + off; off += 8 * 512;
    pk_v[l]  = pkbase + off; off += 8 * 512;
    pk_o[l]  = pkbase + off; off += 8 * 512;
    pk_f1[l] = pkbase + off; off += 24 * 512;
    pk_f2[l] = pkbase + off; off += 24 * 512;
  }

  // pre-pack all weights into WMMA B-fragment images
  k_pack<<<8, 32, 0, stream>>>(cw, DIMS, pk_ctx);
  for (int l = 0; l < 3; ++l) {
    k_pack<<<8, 32, 0, stream>>>(L[l].qw, DIMS, pk_q[l]);
    k_pack<<<8, 32, 0, stream>>>(L[l].kw, DIMS, pk_k[l]);
    k_pack<<<8, 32, 0, stream>>>(L[l].vw, DIMS, pk_v[l]);
    k_pack<<<8, 32, 0, stream>>>(L[l].ow, DIMS, pk_o[l]);
    k_pack<<<24, 32, 0, stream>>>(L[l].f1w, FFN_DIMS, pk_f1[l]);
    k_pack<<<24, 32, 0, stream>>>(L[l].f2w, DIMS, pk_f2[l]);
  }

  k_zero<<<2048, 256, 0, stream>>>(x, (size_t)NPIX * DIMS);
  k_ctx<<<4096, 32, 0, stream>>>(indices, cnts, gene, cemb, pk_ctx, cb, ctx);

  for (int l = 0; l < 3; ++l) {
    k_q<<<1024, 32, 0, stream>>>(x, pk_q[l], L[l].qb, q);
    k_attn<<<NPIX / 16, 128, 0, stream>>>(indptr, ctx, q,
                                          pk_k[l], L[l].kb, pk_v[l], L[l].vb, resb);
    k_oln<<<1024, 32, 0, stream>>>(resb, pk_o[l], L[l].ob, L[l].l1g, L[l].l1b, x);
    k_ffn1<<<dim3(1024, 3), 32, 0, stream>>>(x, pk_f1[l], L[l].f1b, hbuf);
    float* xo = (l == 2) ? (float*)d_out : x;
    k_ffn2<<<1024, 32, 0, stream>>>(hbuf, pk_f2[l], L[l].f2b, L[l].l2g, L[l].l2b, x, xo);
  }
}
